// DynGuidedExtractor_72103910965541
// MI455X (gfx1250) — matrix operand order repaired and checked by reading, not compiled, not run on hardware
//
#include <hip/hip_runtime.h>
#include <hip/hip_bf16.h>

// ---------------------------------------------------------------------------
// Problem constants (from reference): B=8, S=96, H=768, V=16, T=4, K=B*V=128
// ---------------------------------------------------------------------------
#define PB   8
#define PS   96
#define PH   768
#define PH2  1536
#define PG4  3072
#define PV   16
#define PT   4
#define PK   128

typedef __attribute__((ext_vector_type(16))) __bf16 v16bf;
typedef __attribute__((ext_vector_type(8)))  __bf16 v8bf;
typedef __attribute__((ext_vector_type(8)))  float  v8f;

// CDNA5 async global->LDS path via inline asm (ISA §15.18 op 98,
// GLOBAL_LOAD_ASYNC_TO_LDS_B128, tracked by ASYNCcnt). Falls back to a
// reg-bounce ds_store path when the async wait builtin is unavailable
// (e.g. host pass).
#if defined(__AMDGCN__) && __has_builtin(__builtin_amdgcn_s_wait_asynccnt)
#define DGE_ASYNC 1
#else
#define DGE_ASYNC 0
#endif

// ---------------------------------------------------------------------------
// WMMA bf16 fragment helpers (wave32, 16x16x32, f32 accumulate)
// A layout: lane 0-15 = row M, element i -> k = ((i>>3)<<4) + (hi<<3) + (i&7)
// B layout: lane 0-15 = col N, element i -> k = ((i>>3)<<4) + (hi<<3) + (i&7)
// C/D layout: lane holds col N = lane&15, VGPR v holds row M = v + 8*(lane>>4)
// ---------------------------------------------------------------------------
__device__ __forceinline__ v16bf frag_a(const __bf16* r, int hi) {
  v8bf x0 = *reinterpret_cast<const v8bf*>(r + (hi << 3));
  v8bf x1 = *reinterpret_cast<const v8bf*>(r + 16 + (hi << 3));
  v16bf a;
#pragma unroll
  for (int i = 0; i < 8; ++i) { a[i] = x0[i]; a[i + 8] = x1[i]; }
  return a;
}

__device__ __forceinline__ v16bf frag_b(const __bf16* r, int hi) {
  const __bf16* p = r + (hi << 3);
  v8bf x0 = *reinterpret_cast<const v8bf*>(p);
  v8bf x1 = *reinterpret_cast<const v8bf*>(p + 16);
  v16bf b;
#pragma unroll
  for (int i = 0; i < 8; ++i) { b[i] = x0[i]; b[i + 8] = x1[i]; }
  return b;
}

__device__ __forceinline__ v8f wmma_bf16(v16bf a, v16bf b, v8f c) {
  return __builtin_amdgcn_wmma_f32_16x16x32_bf16(
      /*neg_a=*/false, a, /*neg_b=*/false, b,
      /*c_mod=*/(short)0, c, /*reuse_a=*/false, /*reuse_b=*/false);
}

__device__ __forceinline__ float sigmoidf_(float x) {
  return 1.0f / (1.0f + __expf(-x));
}

// 16-byte global -> LDS copy. Generic LDS pointers carry the LDS byte offset
// in addr[31:0] (shared aperture, ISA §10.2), so the truncated pointer is the
// per-lane LDS address the async instruction expects in VDST.
__device__ __forceinline__ void dge_copy16(const __bf16* g, __bf16* l) {
#if DGE_ASYNC
  unsigned loff = (unsigned)(size_t)l;
  unsigned long long ga = (unsigned long long)(size_t)g;
  asm volatile("global_load_async_to_lds_b128 %0, %1, off"
               :: "v"(loff), "v"(ga)
               : "memory");
#else
  *reinterpret_cast<v8bf*>(l) = *reinterpret_cast<const v8bf*>(g);
#endif
}

__device__ __forceinline__ void dge_copy_wait() {
#if DGE_ASYNC
  __builtin_amdgcn_s_wait_asynccnt(0);
#endif
}

// ---------------------------------------------------------------------------
// Stage 1: cls head.  h = relu(cls @ W1^T + b1); scores = h @ W2^T + b2
// One block, h staged in LDS (8*768 floats = 24 KB).
// ---------------------------------------------------------------------------
__global__ void dge_cls_head_kernel(const float* __restrict__ enc_prev,
                                    const float* __restrict__ enc_last,
                                    const float* __restrict__ W1,
                                    const float* __restrict__ b1,
                                    const float* __restrict__ W2,
                                    const float* __restrict__ b2,
                                    float* __restrict__ out_scores) {
  __shared__ float hbuf[PB * PH];
  for (int idx = threadIdx.x; idx < PB * PH; idx += blockDim.x) {
    int b = idx / PH, i = idx % PH;
    const float* w = W1 + (size_t)i * PH2;
    const float* p = enc_prev + (size_t)b * PS * PH;   // token 0
    const float* l = enc_last + (size_t)b * PS * PH;   // token 0
    float acc = b1[i];
    for (int j = 0; j < PH; ++j) acc = fmaf(p[j], w[j], acc);
    for (int j = 0; j < PH; ++j) acc = fmaf(l[j], w[PH + j], acc);
    hbuf[idx] = fmaxf(acc, 0.0f);
  }
  __syncthreads();
  if (threadIdx.x < PB * PV) {
    int b = threadIdx.x / PV, v = threadIdx.x % PV;
    const float* w = W2 + (size_t)v * PH;
    float acc = b2[v];
    for (int j = 0; j < PH; ++j) acc = fmaf(hbuf[b * PH + j], w[j], acc);
    out_scores[threadIdx.x] = acc;
  }
}

// ---------------------------------------------------------------------------
// Stage 2: threshold select (sigmoid(x)>0.5 <=> x>0), row-major scan, pad 0.
// ---------------------------------------------------------------------------
__global__ void dge_select_kernel(const float* __restrict__ scores,
                                  int* __restrict__ idx,      // [256] batch,word
                                  float* __restrict__ valid,  // [128]
                                  float* __restrict__ out_idx) { // d_out+128
  if (threadIdx.x != 0 || blockIdx.x != 0) return;
  int cnt = 0;
  for (int b = 0; b < PB; ++b)
    for (int v = 0; v < PV; ++v)
      if (scores[b * PV + v] > 0.0f) { idx[cnt] = b; idx[PK + cnt] = v; ++cnt; }
  for (int k = cnt; k < PK; ++k) { idx[k] = 0; idx[PK + k] = 0; }
  for (int k = 0; k < PK; ++k) {
    valid[k] = (k < cnt) ? 1.0f : 0.0f;
    out_idx[k]      = (float)idx[k];
    out_idx[PK + k] = (float)idx[PK + k];
  }
}

// ---------------------------------------------------------------------------
// Conversions to bf16
// ---------------------------------------------------------------------------
__global__ void dge_f32_to_bf16_kernel(const float* __restrict__ src,
                                       __bf16* __restrict__ dst, int n) {
  int i = blockIdx.x * blockDim.x + threadIdx.x;
  if (i < n) dst[i] = (__bf16)src[i];
}

// label_emb [V, H*H] row-major (k-major) -> bf16 transposed per word:
// dst[w][j*H + k] = src[w][k*H + j]  (so WMMA B fragments load contiguously)
__global__ void dge_label_transpose_kernel(const float* __restrict__ src,
                                           __bf16* __restrict__ dst) {
  int id = blockIdx.x * blockDim.x + threadIdx.x;
  if (id >= PV * PH * PH) return;
  int w = id / (PH * PH);
  int r = id % (PH * PH);
  int kk = r / PH, j = r % PH;
  dst[(size_t)w * PH * PH + (size_t)j * PH + kk] = (__bf16)src[id];
}

// x[k, s, 0:H] = enc_last_bf16[batch_idx[k], s, :]   (gather, bf16 copy)
__global__ void dge_fill_ctx_kernel(const __bf16* __restrict__ ebf,
                                    const int* __restrict__ idx,
                                    __bf16* __restrict__ xbf) {
  int id = blockIdx.x * blockDim.x + threadIdx.x;
  if (id >= PK * PS * PH) return;
  int k = id / (PS * PH);
  int r = id % (PS * PH);
  int ss = r / PH, j = r % PH;
  int b = idx[k];
  xbf[((size_t)k * PS + ss) * PH2 + j] = ebf[((size_t)b * PS + ss) * PH + j];
}

// ---------------------------------------------------------------------------
// Stage 3: attr_out[k] = ctx[k] (96x768) @ label_emb[w_k] (768x768), bf16 WMMA.
// Grid (K, S/16, H/64), one wave/block computing a 16x64 tile.
// ---------------------------------------------------------------------------
__global__ void __launch_bounds__(32)
dge_attr_gemm_kernel(const __bf16* __restrict__ ebf,
                     const __bf16* __restrict__ lbfT,
                     const int* __restrict__ idx,
                     __bf16* __restrict__ xbf) {
  int k  = blockIdx.x;
  int m0 = blockIdx.y * 16;   // rows within S
  int n0 = blockIdx.z * 64;   // cols within H
  int lane = threadIdx.x;
  int hi = lane >> 4, lr = lane & 15;
  int b = idx[k];
  int w = idx[PK + k];
  const __bf16* Arow = ebf + ((size_t)b * PS + (m0 + lr)) * PH;
  const __bf16* Bm   = lbfT + (size_t)w * PH * PH;

  v8f acc[4] = {};
  for (int kb = 0; kb < PH; kb += 32) {
    v16bf a = frag_a(Arow + kb, hi);
#pragma unroll
    for (int j = 0; j < 4; ++j) {
      const __bf16* Bp = Bm + (size_t)(n0 + j * 16 + lr) * PH + kb;
      acc[j] = wmma_bf16(a, frag_b(Bp, hi), acc[j]);
    }
  }
#pragma unroll
  for (int j = 0; j < 4; ++j) {
    int col = n0 + j * 16 + lr;
#pragma unroll
    for (int v = 0; v < 8; ++v) {
      int row = m0 + v + hi * 8;
      xbf[((size_t)k * PS + row) * PH2 + PH + col] = (__bf16)acc[j][v];
    }
  }
}

// ---------------------------------------------------------------------------
// Zero-init LSTM state (h double-buffered bf16, c fp32)
// ---------------------------------------------------------------------------
__global__ void dge_init_state_kernel(__bf16* __restrict__ hbf,
                                      float* __restrict__ cbuf) {
  int i = blockIdx.x * blockDim.x + threadIdx.x;
  if (i < 4 * PK * PH) hbf[i] = (__bf16)0.0f;   // 2 phase buffers x 2 dirs
  if (i < 2 * PK * PH) cbuf[i] = 0.0f;          // 2 dirs
}

// ---------------------------------------------------------------------------
// Stage 4: one BiLSTM time step (both directions).
//   G = x_t @ W_ih^T + h @ W_hh^T   (K = 1536 + 768, bf16 WMMA, f32 acc)
// Block = 4 waves owning M tiles {0..3}*16 within a 64-row group; all waves
// share the block's hidden-col range n0..n0+63, so the B (weight) chunk
// [4 gates x 64 cols x 32 k] = 16 KB is staged once per block into LDS
// (double-buffered, async global->LDS) and read back with ds_load_b128.
// Grid (K/64, H/64, 2 dirs), block (32,4).
// ---------------------------------------------------------------------------
#define CH_X   (PH2 / 32)          // 48 input-projection chunks
#define CH_ALL (CH_X + PH / 32)    // + 24 recurrent chunks = 72
#define BCH    (256 * 32)          // bf16 elems per LDS chunk buffer (16 KB)

__global__ void __launch_bounds__(128)
dge_lstm_step_kernel(int s,
                     const __bf16* __restrict__ xbf,
                     const __bf16* __restrict__ wih_f, const __bf16* __restrict__ whh_f,
                     const __bf16* __restrict__ wih_b, const __bf16* __restrict__ whh_b,
                     const float* __restrict__ bi_f, const float* __restrict__ bh_f,
                     const float* __restrict__ bi_b, const float* __restrict__ bh_b,
                     __bf16* __restrict__ hbf, float* __restrict__ cbuf,
                     __bf16* __restrict__ hsf, __bf16* __restrict__ hsb) {
  __shared__ __bf16 smem[2 * BCH];                       // 32 KB double buffer

  int dir = blockIdx.z;
  int m0  = (blockIdx.x * 4 + threadIdx.y) * 16;          // batch rows (per wave)
  int n0  = blockIdx.y * 64;                              // hidden cols (per block)
  int lane = threadIdx.x;
  int tid  = threadIdx.y * 32 + threadIdx.x;              // 0..127
  int hi = lane >> 4, lr = lane & 15;
  int t = dir ? (PS - 1 - s) : s;

  const __bf16* wih = dir ? wih_b : wih_f;
  const __bf16* whh = dir ? whh_b : whh_f;
  const float*  bi  = dir ? bi_b  : bi_f;
  const float*  bh  = dir ? bh_b  : bh_f;
  const __bf16* hin  = hbf + ((size_t)((s) & 1) * 2 + dir) * (PK * PH);
  __bf16*       hout = hbf + ((size_t)((s + 1) & 1) * 2 + dir) * (PK * PH);
  float*        c    = cbuf + (size_t)dir * PK * PH;
  __bf16*       hs   = dir ? hsb : hsf;

  const __bf16* ArowX = xbf + ((size_t)(m0 + lr) * PS + t) * PH2;
  const __bf16* HrowH = hin + (size_t)(m0 + lr) * PH;

  // Cooperative B-chunk staging: 256 cols x 32 k bf16, 64 B per col.
  // Thread handles part p (16 B) of col group; 8 passes cover 256 cols.
  int p  = tid & 3;           // which 16-byte slice of a column
  int c0 = tid >> 2;          // base column (0..31)
  auto stage = [&](int ch) {
    const __bf16* w; int ld, kb;
    if (ch < CH_X) { w = wih; ld = PH2; kb = ch * 32; }
    else           { w = whh; ld = PH;  kb = (ch - CH_X) * 32; }
    __bf16* buf = smem + (ch & 1) * BCH;
#pragma unroll
    for (int pass = 0; pass < 8; ++pass) {
      int colIdx = c0 + pass * 32;                 // 0..255
      int g = colIdx >> 6, cl = colIdx & 63;       // gate, col within tile
      const __bf16* gp = w + (size_t)(g * PH + n0 + cl) * ld + kb + p * 8;
      dge_copy16(gp, buf + colIdx * 32 + p * 8);
    }
  };

  v8f acc[4][4] = {};

  stage(0);
  dge_copy_wait();
  __syncthreads();

  for (int ch = 0; ch < CH_ALL; ++ch) {
    if (ch + 1 < CH_ALL) stage(ch + 1);            // overlaps with compute
    const __bf16* ar = (ch < CH_X) ? (ArowX + ch * 32)
                                   : (HrowH + (ch - CH_X) * 32);
    v16bf a = frag_a(ar, hi);
    const __bf16* buf = smem + (ch & 1) * BCH;
#pragma unroll
    for (int g = 0; g < 4; ++g)
#pragma unroll
      for (int j = 0; j < 4; ++j) {
        const __bf16* Bp = buf + (g * 64 + j * 16 + lr) * 32;   // LDS
        acc[g][j] = wmma_bf16(a, frag_b(Bp, hi), acc[g][j]);
      }
    dge_copy_wait();     // own async ops for chunk ch+1 landed
    __syncthreads();     // all waves done reading buf(ch) & staging ch+1
  }

  // fused cell update (pytorch gate order i,f,g,o)
#pragma unroll
  for (int j = 0; j < 4; ++j) {
    int col = n0 + j * 16 + lr;
    float bii = bi[col]          + bh[col];
    float bif = bi[PH + col]     + bh[PH + col];
    float big = bi[2 * PH + col] + bh[2 * PH + col];
    float bio = bi[3 * PH + col] + bh[3 * PH + col];
#pragma unroll
    for (int v = 0; v < 8; ++v) {
      int row = m0 + v + hi * 8;
      size_t ci = (size_t)row * PH + col;
      float ig = sigmoidf_(acc[0][j][v] + bii);
      float fg = sigmoidf_(acc[1][j][v] + bif);
      float gg = tanhf(acc[2][j][v] + big);
      float og = sigmoidf_(acc[3][j][v] + bio);
      float cn = fg * c[ci] + ig * gg;
      c[ci] = cn;
      float hn = og * tanhf(cn);
      __bf16 hv = (__bf16)hn;
      hout[ci] = hv;
      hs[((size_t)row * PS + t) * PH + col] = hv;
    }
  }
}

// ---------------------------------------------------------------------------
// Stage 5: tags = [h_f | h_b] @ W_tag^T + b_tag, masked by valid[k].
// ---------------------------------------------------------------------------
__global__ void dge_tags_kernel(const __bf16* __restrict__ hsf,
                                const __bf16* __restrict__ hsb,
                                const float* __restrict__ W_tag,
                                const float* __restrict__ b_tag,
                                const float* __restrict__ valid,
                                float* __restrict__ tags) {
  int id = blockIdx.x * blockDim.x + threadIdx.x;   // k*S + s
  if (id >= PK * PS) return;
  int k = id / PS;
  float vm = valid[k];
  const __bf16* hf = hsf + (size_t)id * PH;
  const __bf16* hb = hsb + (size_t)id * PH;
  float t0 = 0.f, t1 = 0.f, t2 = 0.f, t3 = 0.f;
  for (int j = 0; j < PH; ++j) {
    float a = (float)hf[j];
    t0 = fmaf(a, W_tag[0 * PH2 + j], t0);
    t1 = fmaf(a, W_tag[1 * PH2 + j], t1);
    t2 = fmaf(a, W_tag[2 * PH2 + j], t2);
    t3 = fmaf(a, W_tag[3 * PH2 + j], t3);
  }
  for (int j = 0; j < PH; ++j) {
    float a = (float)hb[j];
    t0 = fmaf(a, W_tag[0 * PH2 + PH + j], t0);
    t1 = fmaf(a, W_tag[1 * PH2 + PH + j], t1);
    t2 = fmaf(a, W_tag[2 * PH2 + PH + j], t2);
    t3 = fmaf(a, W_tag[3 * PH2 + PH + j], t3);
  }
  tags[(size_t)id * PT + 0] = vm * (t0 + b_tag[0]);
  tags[(size_t)id * PT + 1] = vm * (t1 + b_tag[1]);
  tags[(size_t)id * PT + 2] = vm * (t2 + b_tag[2]);
  tags[(size_t)id * PT + 3] = vm * (t3 + b_tag[3]);
}

// ---------------------------------------------------------------------------
// Host-side launcher
// ---------------------------------------------------------------------------
extern "C" void kernel_launch(void* const* d_in, const int* in_sizes, int n_in,
                              void* d_out, int out_size, void* d_ws, size_t ws_size,
                              hipStream_t stream) {
  (void)in_sizes; (void)n_in; (void)out_size; (void)ws_size;
  const float* enc_prev = (const float*)d_in[0];
  const float* enc_last = (const float*)d_in[1];
  const float* W1       = (const float*)d_in[2];
  const float* b1       = (const float*)d_in[3];
  const float* W2       = (const float*)d_in[4];
  const float* b2       = (const float*)d_in[5];
  const float* label_emb= (const float*)d_in[6];
  const float* W_ih_f   = (const float*)d_in[7];
  const float* W_hh_f   = (const float*)d_in[8];
  const float* b_ih_f   = (const float*)d_in[9];
  const float* b_hh_f   = (const float*)d_in[10];
  const float* W_ih_b   = (const float*)d_in[11];
  const float* W_hh_b   = (const float*)d_in[12];
  const float* b_ih_b   = (const float*)d_in[13];
  const float* b_hh_b   = (const float*)d_in[14];
  const float* W_tag    = (const float*)d_in[15];
  const float* b_tag    = (const float*)d_in[16];

  float* out = (float*)d_out;
  // d_out layout: [0,128) cls_scores | [128,256) batch_idx | [256,384) word_idx
  //               | [384, 384+K*S*T) tags
  float* out_scores = out;
  float* out_idx    = out + PK;
  float* out_tags   = out + 3 * PK;

  char* ws = (char*)d_ws;
  size_t off = 0;
  auto alloc = [&](size_t bytes) -> char* {
    char* r = ws + off;
    off += (bytes + 255) & ~(size_t)255;
    return r;
  };
  int*    idx   = (int*)   alloc(2 * PK * sizeof(int));
  float*  valid = (float*) alloc(PK * sizeof(float));
  __bf16* ebf   = (__bf16*)alloc((size_t)PB * PS * PH * 2);
  __bf16* lbfT  = (__bf16*)alloc((size_t)PV * PH * PH * 2);
  __bf16* wihf  = (__bf16*)alloc((size_t)PG4 * PH2 * 2);
  __bf16* wihb  = (__bf16*)alloc((size_t)PG4 * PH2 * 2);
  __bf16* whhf  = (__bf16*)alloc((size_t)PG4 * PH * 2);
  __bf16* whhb  = (__bf16*)alloc((size_t)PG4 * PH * 2);
  __bf16* xbf   = (__bf16*)alloc((size_t)PK * PS * PH2 * 2);
  __bf16* hsf   = (__bf16*)alloc((size_t)PK * PS * PH * 2);
  __bf16* hsb   = (__bf16*)alloc((size_t)PK * PS * PH * 2);
  __bf16* hbf   = (__bf16*)alloc((size_t)4 * PK * PH * 2);   // 2 phases x 2 dirs
  float*  cbuf  = (float*) alloc((size_t)2 * PK * PH * sizeof(float));

  // Stage 1+2: cls head and selection
  dge_cls_head_kernel<<<1, 256, 0, stream>>>(enc_prev, enc_last, W1, b1, W2, b2,
                                             out_scores);
  dge_select_kernel<<<1, 32, 0, stream>>>(out_scores, idx, valid, out_idx);

  // Conversions
  {
    int n;
    n = PB * PS * PH;
    dge_f32_to_bf16_kernel<<<(n + 255) / 256, 256, 0, stream>>>(enc_last, ebf, n);
    n = PG4 * PH2;
    dge_f32_to_bf16_kernel<<<(n + 255) / 256, 256, 0, stream>>>(W_ih_f, wihf, n);
    dge_f32_to_bf16_kernel<<<(n + 255) / 256, 256, 0, stream>>>(W_ih_b, wihb, n);
    n = PG4 * PH;
    dge_f32_to_bf16_kernel<<<(n + 255) / 256, 256, 0, stream>>>(W_hh_f, whhf, n);
    dge_f32_to_bf16_kernel<<<(n + 255) / 256, 256, 0, stream>>>(W_hh_b, whhb, n);
    n = PV * PH * PH;
    dge_label_transpose_kernel<<<(n + 255) / 256, 256, 0, stream>>>(label_emb, lbfT);
  }

  // Stage 3: gather ctx + batched attr GEMM -> x (bf16 [K,S,2H])
  {
    int n = PK * PS * PH;
    dge_fill_ctx_kernel<<<(n + 255) / 256, 256, 0, stream>>>(ebf, idx, xbf);
  }
  dge_attr_gemm_kernel<<<dim3(PK, PS / 16, PH / 64), 32, 0, stream>>>(ebf, lbfT,
                                                                      idx, xbf);

  // Stage 4: BiLSTM, 96 recurrent steps (both directions per launch)
  {
    int n = 4 * PK * PH;
    dge_init_state_kernel<<<(n + 255) / 256, 256, 0, stream>>>(hbf, cbuf);
  }
  for (int s = 0; s < PS; ++s) {
    dge_lstm_step_kernel<<<dim3(PK / 64, PH / 64, 2), dim3(32, 4), 0, stream>>>(
        s, xbf, wihf, whhf, wihb, whhb,
        b_ih_f, b_hh_f, b_ih_b, b_hh_b,
        hbf, cbuf, hsf, hsb);
  }

  // Stage 5: tags
  {
    int n = PK * PS;
    dge_tags_kernel<<<(n + 127) / 128, 128, 0, stream>>>(hsf, hsb, W_tag, b_tag,
                                                         valid, out_tags);
  }
}
